// SelfAttention_38671885534048
// MI455X (gfx1250) — matrix operand order repaired and checked by reading, not compiled, not run on hardware
//
#include <hip/hip_runtime.h>
#include <hip/hip_bf16.h>

// ---------------- problem constants ----------------
constexpr int Bc   = 2;
constexpr int Sc   = 2048;
constexpr int HIDc = 2048;
constexpr int NHc  = 16;
constexpr int HDc  = 128;

// ---------------- CDNA5 WMMA types ----------------
typedef __bf16 v16bf __attribute__((ext_vector_type(16)));
typedef float  v8f   __attribute__((ext_vector_type(8)));

#define WMMA_BF16(A_, B_, C_) \
  __builtin_amdgcn_wmma_f32_16x16x32_bf16(false, (A_), false, (B_), (short)0, (C_), false, false)

// ---------------- bf16 <-> f32 helpers (RNE) ----------------
__device__ static inline unsigned short f2bf(float f) {
  union { float f; unsigned u; } x; x.f = f;
  unsigned r = x.u + 0x7FFFu + ((x.u >> 16) & 1u);
  return (unsigned short)(r >> 16);
}
__device__ static inline float bf2f(unsigned short s) {
  union { unsigned u; float f; } x; x.u = ((unsigned)s) << 16;
  return x.f;
}

// ---------------- CDNA5 async global->LDS copy (signature from round-2 probe) ----------------
typedef int v4i_gl __attribute__((__vector_size__(16)));

#if __has_builtin(__builtin_amdgcn_global_load_async_to_lds_b128)
#define ASYNC_LDS 1
__device__ static inline void g2l_b128(const unsigned short* g, unsigned short* l) {
  __builtin_amdgcn_global_load_async_to_lds_b128((v4i_gl*)g, (v4i_gl*)l, 0, 0);
}
#if __has_builtin(__builtin_amdgcn_s_wait_asynccnt)
#define WAIT_ASYNC(n) __builtin_amdgcn_s_wait_asynccnt(n)
#else
#define WAIT_ASYNC(n) asm volatile("s_wait_asynccnt %0" ::"i"(n) : "memory")
#endif
#else
#define ASYNC_LDS 0
__device__ static inline void g2l_b128(const unsigned short* g, unsigned short* l) {
  *(uint4*)l = *(const uint4*)g;     // synchronous fallback
}
#define WAIT_ASYNC(n) ((void)0)
#endif

// ---------------- WMMA fragment loaders (wave32, ISA 7.12.2) ----------------
// A (16x32 bf16, row-major, leading dim ld): lane row m=lane&15,
// regs 0..7 -> K = g..g+7 / 16+g..16+g+7 (g = (lane>>4)*8). Contiguous 2x16B.
__device__ static inline v16bf frag_a(const unsigned short* s, int ld) {
  union { v16bf v; unsigned short u[16]; } r;
  const int lane = threadIdx.x & 31;
  const unsigned short* p = s + (lane & 15) * ld + ((lane >> 4) << 3);
#pragma unroll
  for (int e = 0; e < 8; ++e) r.u[e] = p[e];
#pragma unroll
  for (int e = 0; e < 8; ++e) r.u[8 + e] = p[16 + e];
  return r.v;
}

// B via transposed storage: logical B[k][n] = s[n*ld + k]; lane col n=lane&15,
// regs 0..7 -> K = g..g+7 / 16+g..16+g+7 (g = (lane>>4)*8). 2x16B per lane.
__device__ static inline v16bf frag_bT(const unsigned short* s, int ld) {
  union { v16bf v; unsigned short u[16]; } r;
  const int lane = threadIdx.x & 31;
  const unsigned short* p = s + (lane & 15) * ld + ((lane >> 4) << 3);
#pragma unroll
  for (int e = 0; e < 8; ++e) r.u[e] = p[e];
#pragma unroll
  for (int e = 0; e < 8; ++e) r.u[8 + e] = p[16 + e];
  return r.v;
}

// ---------------- Kernel 1: f32 -> bf16 cast ----------------
__global__ __launch_bounds__(256) void cast4(const float4* __restrict__ in,
                                             ushort4* __restrict__ out, int n4) {
  for (int i = blockIdx.x * blockDim.x + threadIdx.x; i < n4;
       i += gridDim.x * blockDim.x) {
    float4 f = in[i];
    ushort4 u;
    u.x = f2bf(f.x); u.y = f2bf(f.y); u.z = f2bf(f.z); u.w = f2bf(f.w);
    out[i] = u;
  }
}

// ---------------- Kernel 1b: f32 [R,C] -> bf16 transposed [C,R] ----------------
__global__ __launch_bounds__(256) void cast_transpose(const float* __restrict__ in,
                                                      unsigned short* __restrict__ out,
                                                      int R, int C) {
  __shared__ unsigned short t[32][33];
  const int bx = blockIdx.x * 32, by = blockIdx.y * 32;
  const int tx = threadIdx.x, ty = threadIdx.y;   // tx 0..31, ty 0..7
#pragma unroll
  for (int j = 0; j < 4; ++j)
    t[ty + j * 8][tx] = f2bf(in[(size_t)(by + ty + j * 8) * C + bx + tx]);
  __syncthreads();
#pragma unroll
  for (int j = 0; j < 4; ++j)
    out[(size_t)(bx + ty + j * 8) * R + by + tx] = t[tx][ty + j * 8];
}

// ---------------- Kernel 2: bf16 WMMA GEMM (double-buffered async pipeline) ----------------
// C = A[MxK] * Bt[NxK]^T + bias. Block tile 128x128, K-slab 32.
// 256 threads = 8 waves (2x4); wave = 64x32 = 4x2 WMMA tiles.
constexpr int BM = 128, BN = 128, BK = 32;

__global__ __launch_bounds__(256) void gemm_bf16_wmma(
    const unsigned short* __restrict__ A,    // [M,K]
    const unsigned short* __restrict__ Bt,   // [N,K]  (transposed weights)
    const float* __restrict__ bias,          // [N]
    float* __restrict__ Cf,
    unsigned short* __restrict__ Cb,
    int M, int N, int K, int store_f32) {
  (void)M;
  __shared__ __align__(16) unsigned short As[2][BM * BK];   // 2 x 8 KB
  __shared__ __align__(16) unsigned short Bs[2][BM * BK];   // 2 x 8 KB

  const int tid = threadIdx.x;
  const int wave = tid >> 5, lane = tid & 31;
  const int wm = wave >> 2, wn = wave & 3;
  const int bm0 = blockIdx.y * BM, bn0 = blockIdx.x * BN;

  // stage one K-slab into buffer bb (4 async b128 per thread; both tiles [128x32])
  auto stage = [&](int bb, int kk) {
#pragma unroll
    for (int i = 0; i < 2; ++i) {
      const int idx = (tid + i * 256) * 8;
      const int rr = idx >> 5, ck = idx & 31;
      g2l_b128(&A[(size_t)(bm0 + rr) * K + kk + ck], &As[bb][idx]);
      g2l_b128(&Bt[(size_t)(bn0 + rr) * K + kk + ck], &Bs[bb][idx]);
    }
  };

  v8f acc[4][2];
#pragma unroll
  for (int mt = 0; mt < 4; ++mt)
#pragma unroll
    for (int nt = 0; nt < 2; ++nt) acc[mt][nt] = {};

  stage(0, 0);                                   // prologue: slab 0 in flight
  int buf = 0;
  for (int k0 = 0; k0 < K; k0 += BK) {
    if (k0 + BK < K) {
      stage(buf ^ 1, k0 + BK);                   // issue next slab first…
      WAIT_ASYNC(4);                             // …then release when slab k lands
    } else {
      WAIT_ASYNC(0);
    }
    if (k0 + 2 * BK < K)                         // far L2 hint (slab k+2)
      __builtin_prefetch(&A[(size_t)(bm0 + (tid >> 1)) * K + k0 + 2 * BK], 0, 1);
    __syncthreads();                             // publish all threads' copies

    v16bf af[4], bf[2];
#pragma unroll
    for (int mt = 0; mt < 4; ++mt)
      af[mt] = frag_a(&As[buf][(wm * 64 + mt * 16) * BK], BK);
#pragma unroll
    for (int nt = 0; nt < 2; ++nt)
      bf[nt] = frag_bT(&Bs[buf][(wn * 32 + nt * 16) * BK], BK);
#pragma unroll
    for (int mt = 0; mt < 4; ++mt)
#pragma unroll
      for (int nt = 0; nt < 2; ++nt)
        acc[mt][nt] = WMMA_BF16(af[mt], bf[nt], acc[mt][nt]);

    __syncthreads();                             // buffer reusable next+1 slab
    buf ^= 1;
  }

  // epilogue: C/D layout = col (lane&15), rows r + 8*(lane>>4)
#pragma unroll
  for (int mt = 0; mt < 4; ++mt) {
#pragma unroll
    for (int nt = 0; nt < 2; ++nt) {
      const int gn = bn0 + wn * 32 + nt * 16 + (lane & 15);
      const int gm0 = bm0 + wm * 64 + mt * 16 + ((lane >> 4) << 3);
      const float bv = bias[gn];
#pragma unroll
      for (int r = 0; r < 8; ++r) {
        float val = acc[mt][nt][r] + bv;
        size_t o = (size_t)(gm0 + r) * N + gn;
        if (store_f32) Cf[o] = val;
        else           Cb[o] = f2bf(val);
      }
    }
  }
}

// ---------------- Kernel 3: RoPE + split + head transpose ----------------
// Also writes V transposed [B,NH,HD,S] so attention can async-stage V directly.
__global__ __launch_bounds__(128) void rope_split(
    const unsigned short* __restrict__ qkv,   // [B*S, NH*384] bf16
    unsigned short* __restrict__ qo,          // [B,NH,S,HD] bf16
    unsigned short* __restrict__ ko,          // [B,NH,S,HD] bf16
    unsigned short* __restrict__ vt,          // [B,NH,HD,S] bf16 (transposed V)
    float* __restrict__ pk,                   // present_k f32 [B,NH,S,HD]
    float* __restrict__ pv) {                 // present_v f32 [B,NH,S,HD]
  const int idx = blockIdx.x;
  const int h = idx % NHc;
  const int bs = idx / NHc;
  const int s = bs % Sc;
  const int b = bs / Sc;
  const int d = threadIdx.x;

  const unsigned short* base = qkv + (size_t)bs * (3 * HIDc) + (size_t)h * (3 * HDc);

  const int i = d & 63;
  float inv_freq = __powf(1.0e-4f, (float)i * (1.0f / 64.0f));
  float ang = (float)s * inv_freq;
  float sn, cs;
  __sincosf(ang, &sn, &cs);

  float ql = bf2f(base[i]);
  float qr = bf2f(base[i + 64]);
  float kl = bf2f(base[128 + i]);
  float kr = bf2f(base[128 + i + 64]);
  float vv = bf2f(base[256 + d]);

  float qv = (d < 64) ? (cs * ql - sn * qr) : (sn * ql + cs * qr);
  float kv = (d < 64) ? (cs * kl - sn * kr) : (sn * kl + cs * kr);

  const size_t o = (((size_t)b * NHc + h) * Sc + s) * HDc + d;
  qo[o] = f2bf(qv);
  ko[o] = f2bf(kv);
  pk[o] = kv;
  pv[o] = vv;
  vt[(((size_t)b * NHc + h) * HDc + d) * Sc + s] = f2bf(vv);
}

// ---------------- Kernel 4: causal flash attention (WMMA, pipelined strips) ----------------
constexpr int BQ = 64;   // query rows per block (4 waves x 16)
constexpr int KT = 32;   // K/V strip width

__global__ __launch_bounds__(128) void attn_fwd(
    const unsigned short* __restrict__ Qg,    // [B,NH,S,HD] bf16 (rope'd)
    const unsigned short* __restrict__ Kg,    // [B,NH,S,HD] bf16 (rope'd)
    const unsigned short* __restrict__ Vt,    // [B,NH,HD,S] bf16 (transposed)
    const float* __restrict__ Mask,           // [B,1,S,S] f32
    unsigned short* __restrict__ Ctx) {       // [B*S, HID] bf16
  __shared__ __align__(16) unsigned short Qs[BQ * HDc];       // 16 KB
  __shared__ __align__(16) unsigned short Ks[2][KT * HDc];    // 2 x 8 KB  [kk][d]
  __shared__ __align__(16) unsigned short Vts[2][HDc * KT];   // 2 x 8 KB  [d][kk]
  __shared__ __align__(16) unsigned short Ps[4][16 * KT];     //  4 KB per-wave P

  const int tid = threadIdx.x;
  const int wave = tid >> 5, lane = tid & 31;
  const int bh = blockIdx.y;
  const int b = bh / NHc, h = bh % NHc;
  const int q0 = blockIdx.x * BQ;

  const size_t hoff = ((size_t)b * NHc + h) * (size_t)Sc * HDc;   // == head off for Vt too
  const unsigned short* Qh = Qg + hoff;
  const unsigned short* Kh = Kg + hoff;
  const unsigned short* Vth = Vt + hoff;
  const float* Mb = Mask + (size_t)b * Sc * Sc;

  // stage a K/V strip into buffer bb (8 async b128 per thread)
  auto stage_kv = [&](int bb, int kk) {
#pragma unroll
    for (int i = 0; i < 4; ++i) {
      const int idx = (tid + i * 128) * 8;
      // K strip [32 x 128]: row kk' = idx>>7, col d = idx&127
      g2l_b128(&Kh[(size_t)(kk + (idx >> 7)) * HDc + (idx & 127)], &Ks[bb][idx]);
      // V strip transposed [128 x 32]: row d = idx>>5, col kk' = idx&31
      g2l_b128(&Vth[(size_t)(idx >> 5) * Sc + kk + (idx & 31)], &Vts[bb][idx]);
    }
  };

  // stage Q tile (64 x 128 bf16) + first K/V strip
#pragma unroll
  for (int i = 0; i < 8; ++i) {
    const int idx = (tid + i * 128) * 8;
    g2l_b128(&Qh[(size_t)(q0 + (idx >> 7)) * HDc + (idx & 127)], &Qs[idx]);
  }
  stage_kv(0, 0);
  WAIT_ASYNC(8);            // Q landed (first 8 ops); strip 0 may still be in flight
  __syncthreads();

  v16bf qf[4];
#pragma unroll
  for (int cc = 0; cc < 4; ++cc)
    qf[cc] = frag_a(&Qs[(wave * 16) * HDc + cc * 32], HDc);

  v8f acc[8];
#pragma unroll
  for (int dt = 0; dt < 8; ++dt) acc[dt] = {};
  float m_i[8], l_i[8];
#pragma unroll
  for (int r = 0; r < 8; ++r) { m_i[r] = -3.0e38f; l_i[r] = 0.0f; }

  const float norm = 0.08838834764831845f;   // 1/sqrt(128)
  const int kend = q0 + BQ;                  // causal bound (uniform per block)

  int buf = 0;
  for (int kb = 0; kb < kend; kb += KT) {
    if (kb + KT < kend) {
      stage_kv(buf ^ 1, kb + KT);            // next strip in flight…
      WAIT_ASYNC(8);                         // …release when current strip lands
    } else {
      WAIT_ASYNC(0);
    }
    __syncthreads();                         // publish all threads' copies

    // scores: S[16x32] = Q[16x128] * K^T[128x32]
    v8f sc[2];
#pragma unroll
    for (int st = 0; st < 2; ++st) {
      sc[st] = {};
#pragma unroll
      for (int cc = 0; cc < 4; ++cc)
        sc[st] = WMMA_BF16(qf[cc], frag_bT(&Ks[buf][(st * 16) * HDc + cc * 32], HDc), sc[st]);
    }

    // online softmax over the 16x32 strip; row = r + 8*(lane>>4), col = lane&15
    const int colbase = kb + (lane & 15);
    const int rowbase = q0 + wave * 16 + ((lane >> 4) << 3);
#pragma unroll
    for (int r = 0; r < 8; ++r) {
      const int qrow = rowbase + r;
      float v0 = sc[0][r] * norm;
      float v1 = sc[1][r] * norm;
      if (colbase > qrow)      v0 = -3.0e38f;
      if (colbase + 16 > qrow) v1 = -3.0e38f;
      v0 += Mb[(size_t)qrow * Sc + colbase];
      v1 += Mb[(size_t)qrow * Sc + colbase + 16];

      float mx = fmaxf(v0, v1);
#pragma unroll
      for (int off = 1; off < 16; off <<= 1)
        mx = fmaxf(mx, __shfl_xor(mx, off, 32));
      float mnew = fmaxf(m_i[r], mx);
      float corr = __expf(m_i[r] - mnew);
      float p0 = __expf(v0 - mnew);
      float p1 = __expf(v1 - mnew);
      float rs = p0 + p1;
#pragma unroll
      for (int off = 1; off < 16; off <<= 1)
        rs += __shfl_xor(rs, off, 32);
      l_i[r] = l_i[r] * corr + rs;
      m_i[r] = mnew;

      const int m = r + ((lane >> 4) << 3);
      Ps[wave][m * KT + (lane & 15)]      = f2bf(p0);
      Ps[wave][m * KT + 16 + (lane & 15)] = f2bf(p1);
#pragma unroll
      for (int dt = 0; dt < 8; ++dt) acc[dt][r] *= corr;
    }
    __syncthreads();   // P re-layout round-trip through LDS

    // ctx += P[16x32] * V[32x128]   (Vts: contiguous B frags)
    v16bf pf = frag_a(Ps[wave], KT);
#pragma unroll
    for (int dt = 0; dt < 8; ++dt)
      acc[dt] = WMMA_BF16(pf, frag_bT(&Vts[buf][(dt * 16) * KT], KT), acc[dt]);

    __syncthreads();   // strip buffer reusable for slab kb+2
    buf ^= 1;
  }

  // normalize and write ctx (layout [b, s, h*HD + d])
#pragma unroll
  for (int r = 0; r < 8; ++r) {
    const float inv = 1.0f / l_i[r];
    const int qrow = q0 + wave * 16 + ((lane >> 4) << 3) + r;
    const size_t ro = ((size_t)b * Sc + qrow) * (size_t)HIDc + (size_t)h * HDc;
#pragma unroll
    for (int dt = 0; dt < 8; ++dt)
      Ctx[ro + dt * 16 + (lane & 15)] = f2bf(acc[dt][r] * inv);
  }
}

// ---------------- launch ----------------
extern "C" void kernel_launch(void* const* d_in, const int* in_sizes, int n_in,
                              void* d_out, int out_size, void* d_ws, size_t ws_size,
                              hipStream_t stream) {
  (void)in_sizes; (void)n_in; (void)out_size; (void)ws_size;

  const float* hs     = (const float*)d_in[0];
  const float* amask  = (const float*)d_in[1];
  const float* wqkv   = (const float*)d_in[2];
  const float* bqkv   = (const float*)d_in[3];
  const float* wdense = (const float*)d_in[4];
  const float* bdense = (const float*)d_in[5];

  float* out = (float*)d_out;
  float* pk  = out + (size_t)Bc * Sc * HIDc;
  float* pv  = pk + (size_t)Bc * NHc * Sc * HDc;

  char* ws = (char*)d_ws;
  size_t o = 0;
  unsigned short* hsb    = (unsigned short*)(ws + o); o += (size_t)Bc * Sc * HIDc * 2;   // hidden bf16, reused as ctx
  unsigned short* wqkvT  = (unsigned short*)(ws + o); o += (size_t)HIDc * 3 * HIDc * 2;  // [3HID, HID]
  unsigned short* wdT    = (unsigned short*)(ws + o); o += (size_t)HIDc * HIDc * 2;      // [HID, HID]
  unsigned short* qkvb   = (unsigned short*)(ws + o); o += (size_t)Bc * Sc * 3 * HIDc * 2;
  unsigned short* qb     = (unsigned short*)(ws + o); o += (size_t)Bc * NHc * Sc * HDc * 2;
  unsigned short* kb     = (unsigned short*)(ws + o); o += (size_t)Bc * NHc * Sc * HDc * 2;
  unsigned short* vtb    = (unsigned short*)(ws + o); o += (size_t)Bc * NHc * HDc * Sc * 2;

  // 1) casts to bf16 (weights transposed for contiguous WMMA B-frags)
  cast4<<<dim3(4096), dim3(256), 0, stream>>>((const float4*)hs, (ushort4*)hsb,
                                              Bc * Sc * HIDc / 4);
  cast_transpose<<<dim3(3 * HIDc / 32, HIDc / 32), dim3(32, 8), 0, stream>>>(
      wqkv, wqkvT, HIDc, 3 * HIDc);
  cast_transpose<<<dim3(HIDc / 32, HIDc / 32), dim3(32, 8), 0, stream>>>(
      wdense, wdT, HIDc, HIDc);

  // 2) QKV projection -> bf16
  gemm_bf16_wmma<<<dim3(3 * HIDc / BN, Bc * Sc / BM), dim3(256), 0, stream>>>(
      hsb, wqkvT, bqkv, nullptr, qkvb, Bc * Sc, 3 * HIDc, HIDc, 0);

  // 3) RoPE + head split/transpose (+ present_k / present_v, + V transposed)
  rope_split<<<dim3(Bc * Sc * NHc), dim3(128), 0, stream>>>(qkvb, qb, kb, vtb, pk, pv);

  // 4) causal flash attention -> ctx bf16 (reuses hsb)
  attn_fwd<<<dim3(Sc / BQ, Bc * NHc), dim3(128), 0, stream>>>(qb, kb, vtb, amask, hsb);

  // 5) dense projection -> f32 out
  gemm_bf16_wmma<<<dim3(HIDc / BN, Bc * Sc / BM), dim3(256), 0, stream>>>(
      hsb, wdT, bdense, out, nullptr, Bc * Sc, HIDc, HIDc, 1);
}
